// BiLSTM_24567212934024
// MI455X (gfx1250) — compile-verified, matrix operand order checked by reading, not run on hardware
//
#include <hip/hip_runtime.h>
#include <hip/hip_bf16.h>

// ---------------------------------------------------------------------------
// BiLSTM for MI455X (gfx1250): bf16 WMMA recurrence, TDM-staged A operands,
// fused gate update. Reference quirks honored: backward cell uses FORWARD
// weights; out_b is out_f reversed in time.
// ---------------------------------------------------------------------------

constexpr int B = 64;
constexpr int T = 512;
constexpr int I = 1024;
constexpr int H = 1024;
constexpr int G = 4 * H;          // 4096 gate columns (i,f,g,o)
constexpr int LD = 1024;          // row stride for x-slab / h (I == H)
constexpr unsigned SLAB_BYTES = (unsigned)B * LD * 2;      // 128 KB per A slab
constexpr unsigned DYN_LDS    = 2u * SLAB_BYTES;           // x + h slabs

typedef __attribute__((ext_vector_type(16))) __bf16 v16bf;
typedef __attribute__((ext_vector_type(8)))  __bf16 v8bf;
typedef __attribute__((ext_vector_type(8)))  float  v8f;
typedef __attribute__((ext_vector_type(4)))  unsigned v4u;
typedef __attribute__((ext_vector_type(8)))  int      v8i;
typedef __attribute__((ext_vector_type(4)))  int      v4i;

union Frag16 { v16bf v; v8bf h[2]; };

__device__ __forceinline__ __bf16 f2bf(float f) {
  unsigned u = __builtin_bit_cast(unsigned, f);
  unsigned r = u + 0x7FFFu + ((u >> 16) & 1u);   // round-to-nearest-even
  unsigned short s = (unsigned short)(r >> 16);
  return __builtin_bit_cast(__bf16, s);
}

__device__ __forceinline__ float sigmoidf(float x) {
  return 1.0f / (1.0f + __expf(-x));
}

// ---------------------------------------------------------------------------
// Tensor Data Mover: async 2-D tile load (rows x cols bf16, row-major) into
// LDS. Descriptor bitfields per CDNA5 ISA §8.3/§8.4. One call per wave.
// ---------------------------------------------------------------------------
__device__ __forceinline__ void tdm_load_2d_bf16(unsigned lds_byte_off,
                                                 const void* gsrc,
                                                 unsigned rows,
                                                 unsigned cols,
                                                 unsigned stride_elems) {
  unsigned long long ga = (unsigned long long)gsrc;
  v4u g0;
  g0[0] = 0x1u;                                   // count=1: valid descriptor
  g0[1] = lds_byte_off;                           // lds_addr
  g0[2] = (unsigned)(ga & 0xFFFFFFFFu);           // global_addr[31:0]
  g0[3] = (unsigned)((ga >> 32) & 0x01FFFFFFu)    // global_addr[56:32]
        | (2u << 30);                             // type = 2 ("image")
  v8i g1;
  g1[0] = (int)(1u << 16);                        // data_size=1 (2 bytes)
  g1[1] = (int)((cols & 0xFFFFu) << 16);          // tensor_dim0[15:0]
  g1[2] = (int)(((cols >> 16) & 0xFFFFu)          // tensor_dim0[31:16]
              | ((rows & 0xFFFFu) << 16));        // tensor_dim1[15:0]
  g1[3] = (int)(((rows >> 16) & 0xFFFFu)          // tensor_dim1[31:16]
              | ((cols & 0xFFFFu) << 16));        // tile_dim0 = cols
  g1[4] = (int)(rows & 0xFFFFu);                  // tile_dim1 = rows; tile_dim2=0
  g1[5] = (int)stride_elems;                      // tensor_dim0_stride[31:0]
  g1[6] = 0;                                      // stride[47:32] | dim1_stride lo
  g1[7] = 0;
  v4i z4 = {0, 0, 0, 0};
#if defined(__clang_major__) && (__clang_major__ >= 23)
  v8i z8 = {0, 0, 0, 0, 0, 0, 0, 0};
  __builtin_amdgcn_tensor_load_to_lds(g0, g1, z4, z4, z8, 0);
#else
  __builtin_amdgcn_tensor_load_to_lds(g0, g1, z4, z4, 0);
#endif
}

// ---------------------------------------------------------------------------
// x (B,T,I) fp32  ->  xbf (T,B,I) bf16   (timestep-contiguous for the GEMM)
// ---------------------------------------------------------------------------
__global__ void cvt_x_kernel(const float* __restrict__ x,
                             __bf16* __restrict__ xbf) {
  size_t idx = ((size_t)blockIdx.x * blockDim.x + threadIdx.x) * 4;
  size_t i  = idx % I;
  size_t bt = idx / I;
  size_t tt = bt % T;
  size_t bb = bt / T;
  const float4 v = *(const float4*)(x + idx);
  __bf16* dst = xbf + ((tt * B + bb) * (size_t)I + i);
  dst[0] = f2bf(v.x); dst[1] = f2bf(v.y); dst[2] = f2bf(v.z); dst[3] = f2bf(v.w);
}

// ---------------------------------------------------------------------------
// Weights fp32 -> bf16, bias = bih_f + bhh_f
// ---------------------------------------------------------------------------
__global__ void cvt_w_kernel(const float* __restrict__ Wih,
                             const float* __restrict__ Whh,
                             const float* __restrict__ bih,
                             const float* __restrict__ bhh,
                             __bf16* __restrict__ Wihb,
                             __bf16* __restrict__ Whhb,
                             float* __restrict__ bias) {
  size_t idx = ((size_t)blockIdx.x * blockDim.x + threadIdx.x) * 4;
  float4 a = *(const float4*)(Wih + idx);
  float4 b = *(const float4*)(Whh + idx);
  Wihb[idx + 0] = f2bf(a.x); Wihb[idx + 1] = f2bf(a.y);
  Wihb[idx + 2] = f2bf(a.z); Wihb[idx + 3] = f2bf(a.w);
  Whhb[idx + 0] = f2bf(b.x); Whhb[idx + 1] = f2bf(b.y);
  Whhb[idx + 2] = f2bf(b.z); Whhb[idx + 3] = f2bf(b.w);
  if (idx < (size_t)G) {
    bias[idx + 0] = bih[idx + 0] + bhh[idx + 0];
    bias[idx + 1] = bih[idx + 1] + bhh[idx + 1];
    bias[idx + 2] = bih[idx + 2] + bhh[idx + 2];
    bias[idx + 3] = bih[idx + 3] + bhh[idx + 3];
  }
}

// ---------------------------------------------------------------------------
// Initial states
// ---------------------------------------------------------------------------
__global__ void init_state_kernel(const float* __restrict__ h0f,
                                  const float* __restrict__ c0f,
                                  const float* __restrict__ h0b,
                                  const float* __restrict__ c0b,
                                  __bf16* __restrict__ hfb0,
                                  __bf16* __restrict__ hbb0,
                                  float* __restrict__ cf,
                                  float* __restrict__ cb,
                                  float* __restrict__ hf32,
                                  float* __restrict__ hb32) {
  size_t i = (size_t)blockIdx.x * blockDim.x + threadIdx.x;
  float hf = h0f[i], hb = h0b[i];
  hfb0[i] = f2bf(hf); hbb0[i] = f2bf(hb);
  cf[i] = c0f[i];     cb[i] = c0b[i];
  hf32[i] = hf;       hb32[i] = hb;
}

// ---------------------------------------------------------------------------
// One LSTM time step, both directions (64 blocks = 2 dirs x 32 col-groups).
// A operands (x_t and h slabs, 64x1024 bf16 each) are DMA'd into LDS by the
// Tensor Data Mover once per block; the WMMA K-loop reads A from LDS and
// streams the per-block weight slice from L2. Block owns hidden units
// [hc, hc+32) across all four gates, so the cell update fuses in-block.
// ---------------------------------------------------------------------------
__global__ __launch_bounds__(256) void lstm_step_kernel(
    const __bf16* __restrict__ xbf,
    const __bf16* __restrict__ Wih,
    const __bf16* __restrict__ Whh,
    const float*  __restrict__ bias,
    const __bf16* __restrict__ hf_in, __bf16* __restrict__ hf_out,
    const __bf16* __restrict__ hb_in, __bf16* __restrict__ hb_out,
    float* __restrict__ cf, float* __restrict__ cb,
    float* __restrict__ hf32, float* __restrict__ hb32,
    float* __restrict__ out, int t) {
  __shared__ float ldsG[4][64][33];   // [gate][batch row][hidden col], padded
  extern __shared__ __bf16 dynA[];    // [0,64K): x slab, [64K,128K): h slab (elems)

  const int dir     = blockIdx.x >> 5;          // 0 = fwd, 1 = bwd
  const int hcBlock = (blockIdx.x & 31) * 32;   // hidden column group
  const int t_eff   = dir ? (T - 1 - t) : t;

  const __bf16* xslab = xbf + (size_t)t_eff * B * I;  // 64 x 1024 bf16, dense
  const __bf16* hin   = dir ? hb_in  : hf_in;         // 64 x 1024 bf16, dense
  __bf16*       hout  = dir ? hb_out : hf_out;
  float*        cst   = dir ? cb     : cf;
  float*        h32   = dir ? hb32   : hf32;

  const int lane  = threadIdx.x & 31;
  const int wave  = threadIdx.x >> 5;   // 8 waves
  const int gate  = wave >> 1;          // 0..3 -> i,f,g,o
  const int nhalf = wave & 1;           // low/high 16 hidden cols
  const int l16   = lane & 15;
  const int khi   = (lane >> 4) * 8;    // K sub-offset per ISA 16-bit layout
  const int nglob = gate * H + hcBlock + nhalf * 16 + l16; // weight row

  // --- TDM: stage both A slabs into LDS (wave 0 only; TDM ignores EXEC) ----
  if (wave == 0) {
    const unsigned dynBase = __builtin_amdgcn_groupstaticsize();
    tdm_load_2d_bf16(dynBase,              xslab, B, LD, LD);
    tdm_load_2d_bf16(dynBase + SLAB_BYTES, hin,   B, LD, LD);
    __builtin_amdgcn_s_wait_tensorcnt(0);
  }
  __syncthreads();

  const __bf16* ldsX = dynA;
  const __bf16* ldsH = dynA + (size_t)B * LD;

  const __bf16* Wrow_i = Wih + (size_t)nglob * I;
  const __bf16* Wrow_h = Whh + (size_t)nglob * H;

  v8f acc[4] = {v8f{}, v8f{}, v8f{}, v8f{}};

  // K over input projection (x_t @ Wih^T), A from LDS, B from global/L2
#pragma unroll 2
  for (int k0 = 0; k0 < I; k0 += 32) {
    const __bf16* wb = Wrow_i + k0 + khi;
    Frag16 bfr;
    bfr.h[0] = *(const v8bf*)(wb);
    bfr.h[1] = *(const v8bf*)(wb + 16);
    __builtin_prefetch(wb + 32, 0, 1);   // global_prefetch_b8 on weight stream
    const __bf16* abase = ldsX + k0 + khi;
#pragma unroll
    for (int m = 0; m < 4; ++m) {
      const __bf16* ar = abase + (size_t)(m * 16 + l16) * LD;
      Frag16 afr;
      afr.h[0] = *(const v8bf*)(ar);
      afr.h[1] = *(const v8bf*)(ar + 16);
      acc[m] = __builtin_amdgcn_wmma_f32_16x16x32_bf16(
          false, afr.v, false, bfr.v, (short)0, acc[m], false, false);
    }
  }
  // K over recurrence (h @ Whh^T)
#pragma unroll 2
  for (int k0 = 0; k0 < H; k0 += 32) {
    const __bf16* wb = Wrow_h + k0 + khi;
    Frag16 bfr;
    bfr.h[0] = *(const v8bf*)(wb);
    bfr.h[1] = *(const v8bf*)(wb + 16);
    __builtin_prefetch(wb + 32, 0, 1);
    const __bf16* abase = ldsH + k0 + khi;
#pragma unroll
    for (int m = 0; m < 4; ++m) {
      const __bf16* ar = abase + (size_t)(m * 16 + l16) * LD;
      Frag16 afr;
      afr.h[0] = *(const v8bf*)(ar);
      afr.h[1] = *(const v8bf*)(ar + 16);
      acc[m] = __builtin_amdgcn_wmma_f32_16x16x32_bf16(
          false, afr.v, false, bfr.v, (short)0, acc[m], false, false);
    }
  }

  // Stage gate pre-activations (+bias) to LDS per the 16x16 f32 C/D layout:
  // VGPR r at lane L -> row m*16 + r + (L<16?0:8), col (L&15).
  const float bv     = bias[nglob];
  const int   colL   = nhalf * 16 + l16;
  const int   rowOff = (lane >> 4) * 8;
#pragma unroll
  for (int m = 0; m < 4; ++m)
#pragma unroll
    for (int r = 0; r < 8; ++r)
      ldsG[gate][m * 16 + rowOff + r][colL] = acc[m][r] + bv;

  __syncthreads();

  // Elementwise cell update: 64 batch rows x 32 hidden cols = 2048 elems.
#pragma unroll
  for (int e = 0; e < 8; ++e) {
    int p    = e * 256 + threadIdx.x;
    int brow = p >> 5;
    int col  = p & 31;
    int hid  = hcBlock + col;
    float iv = sigmoidf(ldsG[0][brow][col]);
    float fv = sigmoidf(ldsG[1][brow][col]);
    float gv = tanhf   (ldsG[2][brow][col]);
    float ov = sigmoidf(ldsG[3][brow][col]);
    size_t ci = (size_t)brow * H + hid;
    float cn = fv * cst[ci] + iv * gv;
    float hn = ov * tanhf(cn);
    cst[ci]  = cn;
    hout[ci] = f2bf(hn);
    h32[ci]  = hn;
    if (dir == 0) {
      // out_f[t] -> output[:, t, 0:H]; out_b = out_f reversed -> [:, T-1-t, H:2H]
      size_t ob = (size_t)brow * T * (2 * H);
      out[ob + (size_t)t * (2 * H) + hid]               = hn;
      out[ob + (size_t)(T - 1 - t) * (2 * H) + H + hid] = hn;
    }
  }
}

// ---------------------------------------------------------------------------
// Final (h_i, c_i) tails of d_out
// ---------------------------------------------------------------------------
__global__ void finalize_kernel(const float* __restrict__ hf32,
                                const float* __restrict__ hb32,
                                const float* __restrict__ cf,
                                const float* __restrict__ cb,
                                float* __restrict__ out_states) {
  size_t i = (size_t)blockIdx.x * blockDim.x + threadIdx.x;  // over B*2H
  size_t b = i / (2 * H);
  size_t j = i % (2 * H);
  float h = (j < H) ? hf32[b * H + j] : hb32[b * H + (j - H)];
  float c = (j < H) ? cf[b * H + j]   : cb[b * H + (j - H)];
  out_states[i]                     = h;
  out_states[(size_t)B * 2 * H + i] = c;
}

// ---------------------------------------------------------------------------
extern "C" void kernel_launch(void* const* d_in, const int* in_sizes, int n_in,
                              void* d_out, int out_size, void* d_ws, size_t ws_size,
                              hipStream_t stream) {
  const float* x     = (const float*)d_in[0];
  const float* Wih_f = (const float*)d_in[1];
  const float* Whh_f = (const float*)d_in[2];
  const float* bih_f = (const float*)d_in[3];
  const float* bhh_f = (const float*)d_in[4];
  // d_in[5..8] (backward weights/biases) are unused: the reference applies
  // the forward parameter set to both directions.
  const float* h0f   = (const float*)d_in[9];
  const float* c0f   = (const float*)d_in[10];
  const float* h0b   = (const float*)d_in[11];
  const float* c0b   = (const float*)d_in[12];
  float* out = (float*)d_out;

  // Workspace carve-up (~82 MB total)
  char* ws = (char*)d_ws;
  size_t off = 0;
  auto carve = [&](size_t bytes) -> void* {
    void* p = ws + off;
    off = (off + bytes + 255) & ~(size_t)255;
    return p;
  };
  __bf16* xbf   = (__bf16*)carve((size_t)T * B * I * sizeof(__bf16)); // 64 MB
  __bf16* Wihb  = (__bf16*)carve((size_t)G * I * sizeof(__bf16));     // 8 MB
  __bf16* Whhb  = (__bf16*)carve((size_t)G * H * sizeof(__bf16));     // 8 MB
  float*  bias  = (float*) carve((size_t)G * sizeof(float));
  __bf16* hfb0  = (__bf16*)carve((size_t)B * H * sizeof(__bf16));
  __bf16* hfb1  = (__bf16*)carve((size_t)B * H * sizeof(__bf16));
  __bf16* hbb0  = (__bf16*)carve((size_t)B * H * sizeof(__bf16));
  __bf16* hbb1  = (__bf16*)carve((size_t)B * H * sizeof(__bf16));
  float*  cf    = (float*) carve((size_t)B * H * sizeof(float));
  float*  cb    = (float*) carve((size_t)B * H * sizeof(float));
  float*  hf32  = (float*) carve((size_t)B * H * sizeof(float));
  float*  hb32  = (float*) carve((size_t)B * H * sizeof(float));
  (void)ws_size; (void)in_sizes; (void)n_in; (void)out_size;

  // 1) conversions
  {
    size_t n = (size_t)B * T * I / 4;   // 4 elems/thread
    cvt_x_kernel<<<dim3((unsigned)(n / 256)), dim3(256), 0, stream>>>(x, xbf);
  }
  {
    size_t n = (size_t)G * I / 4;
    cvt_w_kernel<<<dim3((unsigned)(n / 256)), dim3(256), 0, stream>>>(
        Wih_f, Whh_f, bih_f, bhh_f, Wihb, Whhb, bias);
  }
  init_state_kernel<<<dim3((B * H) / 256), dim3(256), 0, stream>>>(
      h0f, c0f, h0b, c0b, hfb0, hbb0, cf, cb, hf32, hb32);

  // 2) sequential recurrence: 512 step launches, h state ping-pong,
  //    256 KB dynamic LDS per block for TDM-staged A slabs.
  __bf16* hfbuf[2] = {hfb0, hfb1};
  __bf16* hbbuf[2] = {hbb0, hbb1};
  for (int t = 0; t < T; ++t) {
    int pin = t & 1, pout = pin ^ 1;
    lstm_step_kernel<<<dim3(64), dim3(256), DYN_LDS, stream>>>(
        xbf, Wihb, Whhb, bias,
        hfbuf[pin], hfbuf[pout], hbbuf[pin], hbbuf[pout],
        cf, cb, hf32, hb32, out, t);
  }

  // 3) (h_i, c_i)
  float* out_states = out + (size_t)B * T * 2 * H;
  finalize_kernel<<<dim3((B * 2 * H) / 256), dim3(256), 0, stream>>>(
      hf32, hb32, cf, cb, out_states);
}
